// LTAE_11433202942213
// MI455X (gfx1250) — matrix operand order repaired and checked by reading, not compiled, not run on hardware
//
#include <hip/hip_runtime.h>
#include <hip/hip_bf16.h>
#include <stdint.h>

typedef __attribute__((ext_vector_type(16))) __bf16 v16bf;
typedef __attribute__((ext_vector_type(8)))  __bf16 v8bf;
typedef __attribute__((ext_vector_type(8)))  float  v8f;

// Problem constants
constexpr int B_  = 2048, T_ = 64, C_ = 1024, H_ = 16, NK_ = 128; // NK = H*DK
constexpr int M1_ = 512,  M2_ = 128;
constexpr float SCALE_ = 0.35355339059327373f;   // 1/sqrt(8)
constexpr float EPS_   = 1e-5f;

// LDS row strides (bf16 elems); 16B-aligned rows, rotate banks per row
constexpr int XSTR = 1032;   // for K=1024 tiles
constexpr int HSTR = 520;    // for K=512 tiles

// Workspace layout (bytes)
constexpr size_t OFF_WKT  = 0;                                    // [128][1024] u16
constexpr size_t OFF_W1T  = OFF_WKT + (size_t)NK_ * C_ * 2;       // [512][1024] u16
constexpr size_t OFF_W2T  = OFF_W1T + (size_t)M1_ * C_ * 2;       // [128][512]  u16
constexpr size_t OFF_POOL = OFF_W2T + (size_t)M2_ * M1_ * 2;      // [B][1024] f32
constexpr size_t OFF_H    = OFF_POOL + (size_t)B_ * C_ * 4;       // [B][512] u16

// ---------- helpers ----------
__device__ inline unsigned short f2bf(float f) {           // RNE f32 -> bf16 bits
  unsigned u = __builtin_bit_cast(unsigned, f);
  u += 0x7fffu + ((u >> 16) & 1u);
  return (unsigned short)(u >> 16);
}
__device__ inline unsigned pk2bf(float a, float b) {       // two bf16 in a dword
  return (unsigned)f2bf(a) | ((unsigned)f2bf(b) << 16);
}
__device__ inline float bf2f(unsigned short s) {
  return __builtin_bit_cast(float, (unsigned)s << 16);
}
__device__ inline v8bf ld8(const unsigned short* p) {      // 16B vector load
  return *reinterpret_cast<const v8bf*>(p);
}
__device__ inline v16bf cat16(v8bf lo, v8bf hi) {
  v16bf r;
#pragma unroll
  for (int i = 0; i < 8; ++i) { r[i] = lo[i]; r[i + 8] = hi[i]; }
  return r;
}
__device__ inline v8f wmma_bf16(v16bf a, v16bf b, v8f c) {
  return __builtin_amdgcn_wmma_f32_16x16x32_bf16(false, a, false, b, (short)0, c,
                                                 false, false);
}
__device__ inline float wred_max(float v) {
#pragma unroll
  for (int m = 16; m; m >>= 1) v = fmaxf(v, __shfl_xor(v, m, 32));
  return v;
}
__device__ inline float wred_sum(float v) {
#pragma unroll
  for (int m = 16; m; m >>= 1) v += __shfl_xor(v, m, 32);
  return v;
}

// Fully-unrolled ping-pong WMMA GEMM strip. After unrolling the ping-pong
// buffers are SSA values (no loop-carried copies -> no WMMA hazard NOPs),
// and the scheduler can hoist the global B loads across "iterations".
//  ap0: LDS A base for this wave row (= tile + (mt*16+ln)*ASTR + 16*hf); kk step = 32
//  b0 : global B base (= wT + (ntbase*16+ln)*KD + 16*hf); j step = 16*KD elems
template <int NJ, int KS, int KD>
__device__ __forceinline__ void gemm_pp(const unsigned short* __restrict__ ap0,
                                        const unsigned short* __restrict__ b0,
                                        v8f* acc) {
  v8bf lo0[NJ], hi0[NJ], lo1[NJ], hi1[NJ];
#pragma unroll
  for (int j = 0; j < NJ; ++j) {          // preload kk = 0
    lo0[j] = ld8(b0 + j * 16 * KD);
    hi0[j] = ld8(b0 + j * 16 * KD + 8);
  }
#pragma unroll
  for (int kk = 0; kk < KS; kk += 2) {
    v16bf A0 = cat16(ld8(ap0 + kk * 32), ld8(ap0 + kk * 32 + 8));
#pragma unroll
    for (int j = 0; j < NJ; ++j) {        // prefetch kk+1 while kk computes
      lo1[j] = ld8(b0 + j * 16 * KD + (kk + 1) * 32);
      hi1[j] = ld8(b0 + j * 16 * KD + (kk + 1) * 32 + 8);
    }
#pragma unroll
    for (int j = 0; j < NJ; ++j)
      acc[j] = wmma_bf16(A0, cat16(lo0[j], hi0[j]), acc[j]);

    v16bf A1 = cat16(ld8(ap0 + (kk + 1) * 32), ld8(ap0 + (kk + 1) * 32 + 8));
    if (kk + 2 < KS) {                    // folds away at the last iteration
#pragma unroll
      for (int j = 0; j < NJ; ++j) {      // prefetch kk+2 while kk+1 computes
        lo0[j] = ld8(b0 + j * 16 * KD + (kk + 2) * 32);
        hi0[j] = ld8(b0 + j * 16 * KD + (kk + 2) * 32 + 8);
      }
    }
#pragma unroll
    for (int j = 0; j < NJ; ++j)
      acc[j] = wmma_bf16(A1, cat16(lo1[j], hi1[j]), acc[j]);
  }
}

// ---------- kernel 0: convert weights fp32 -> bf16, transposed to [N][K] ----------
__global__ void __launch_bounds__(256) k_prep(const float* __restrict__ Wk,
                                              const float* __restrict__ W1,
                                              const float* __restrict__ W2,
                                              unsigned short* __restrict__ wkT,
                                              unsigned short* __restrict__ w1T,
                                              unsigned short* __restrict__ w2T) {
  const int nthr = gridDim.x * blockDim.x;
  const int gid  = blockIdx.x * blockDim.x + threadIdx.x;
  for (int i = gid; i < C_ * NK_; i += nthr) {           // Wk (1024 x 128)
    int c = i >> 7, n = i & 127;
    wkT[n * C_ + c] = f2bf(Wk[i]);
  }
  for (int i = gid; i < C_ * M1_; i += nthr) {           // W1 (1024 x 512)
    int c = i >> 9, n = i & 511;
    w1T[n * C_ + c] = f2bf(W1[i]);
  }
  for (int i = gid; i < M1_ * M2_; i += nthr) {          // W2 (512 x 128)
    int c = i >> 7, n = i & 127;
    w2T[n * M1_ + c] = f2bf(W2[i]);
  }
}

// ---------- kernel 1: fused attention per batch element ----------
// LDS: xb[64][XSTR] bf16 (129KB) | kmat[64][128] f32 (32KB) | at[16][64] f32 (4KB)
__global__ void __launch_bounds__(256) k_attn(
    const float* __restrict__ x, const uint8_t* __restrict__ kpm,
    const float* __restrict__ bk, const float* __restrict__ q,
    const unsigned short* __restrict__ wkT,
    float* __restrict__ pool, float* __restrict__ attn_out) {
  extern __shared__ char smem[];
  unsigned short* xb   = (unsigned short*)smem;            // [64][XSTR]
  float*          kmat = (float*)(smem + (size_t)T_ * XSTR * 2);
  float*          at   = kmat + T_ * NK_;

  const int b    = blockIdx.x;
  const int tid  = threadIdx.x;
  const int wv   = tid >> 5;
  const int lane = tid & 31, ln = lane & 15, hf = lane >> 4;

  // ---- stage x[b] -> bf16 LDS (single HBM pass over x) ----
  const float* xrow = x + (size_t)b * T_ * C_;
#pragma unroll 4
  for (int i = 0; i < 64; ++i) {
    int id = tid + 256 * i;                 // float4 chunk id; 256 chunks/row
    int r = id >> 8, c4 = (id & 255) << 2;
    const float4 v = *reinterpret_cast<const float4*>(xrow + r * C_ + c4);
    uint2 p{pk2bf(v.x, v.y), pk2bf(v.z, v.w)};
    *reinterpret_cast<uint2*>(xb + r * XSTR + c4) = p;
  }
  __syncthreads();

  // ---- k = xb @ Wk (+bk) via bf16 WMMA; M=64, N=128, K=1024 ----
  {
    const int mt = wv & 3, ntbase = (wv >> 2) * 4;
    v8f acc[4] = {};
    gemm_pp<4, 32, C_>(xb + (mt * 16 + ln) * XSTR + 16 * hf,
                       wkT + (ntbase * 16 + ln) * C_ + 16 * hf, acc);
#pragma unroll
    for (int j = 0; j < 4; ++j) {
      int n = (ntbase + j) * 16 + ln;
      float bias = bk[n];
#pragma unroll
      for (int r = 0; r < 8; ++r)       // D: lane col = ln, row = r + 8*hf
        kmat[(mt * 16 + r + 8 * hf) * NK_ + n] = acc[j][r] + bias;
    }
  }
  __syncthreads();

  // ---- scores + masked softmax (2 heads per wave, 2 t per lane) ----
  const uint8_t* kb = kpm + (size_t)b * T_;
  float* aout = attn_out + (size_t)b * H_ * T_;
#pragma unroll
  for (int hh = 0; hh < 2; ++hh) {
    int h = wv * 2 + hh;
    float qv[8];
#pragma unroll
    for (int d = 0; d < 8; ++d) qv[d] = q[h * 8 + d] * SCALE_;
    int t0 = lane, t1 = lane + 32;
    float s0 = 0.f, s1 = 0.f;
#pragma unroll
    for (int d = 0; d < 8; ++d) {
      s0 += qv[d] * kmat[t0 * NK_ + h * 8 + d];
      s1 += qv[d] * kmat[t1 * NK_ + h * 8 + d];
    }
    float mx = wred_max(fmaxf(s0, s1));         // raw max (shift-invariant)
    bool m0 = kb[t0] != 0, m1 = kb[t1] != 0;
    float e0 = m0 ? 0.f : __expf(s0 - mx);
    float e1 = m1 ? 0.f : __expf(s1 - mx);
    float inv = 1.0f / wred_sum(e0 + e1);
    float a0 = e0 * inv, a1 = e1 * inv;
    at[h * T_ + t0] = a0;  at[h * T_ + t1] = a1;
    aout[h * T_ + t0] = a0; aout[h * T_ + t1] = a1;
  }
  __syncthreads();

  // ---- pooled[b, c] = sum_t attn[h(c), t] * x[b, t, c], re-using LDS tile ----
  {
    int c0 = tid * 4;            // 4 channels/thread, all within one head block
    int h  = c0 >> 6;
    float a0 = 0.f, a1 = 0.f, a2 = 0.f, a3 = 0.f;
    for (int t = 0; t < T_; ++t) {
      float w = at[h * T_ + t];
      uint2 u = *reinterpret_cast<const uint2*>(xb + t * XSTR + c0);
      a0 += w * bf2f((unsigned short)(u.x & 0xffffu));
      a1 += w * bf2f((unsigned short)(u.x >> 16));
      a2 += w * bf2f((unsigned short)(u.y & 0xffffu));
      a3 += w * bf2f((unsigned short)(u.y >> 16));
    }
    *reinterpret_cast<float4*>(pool + (size_t)b * C_ + c0) = float4{a0, a1, a2, a3};
  }
}

// ---------- kernel 2: h = relu(LN(pool @ W1)) -> bf16 ----------
// 32 rows/block. LDS: a_t[32][XSTR] bf16 (64.5KB) | ymat[32][512] f32 (64KB)
__global__ void __launch_bounds__(256) k_mlp1(
    const float* __restrict__ pool, const unsigned short* __restrict__ w1T,
    const float* __restrict__ g1, const float* __restrict__ b1,
    unsigned short* __restrict__ hout) {
  extern __shared__ char smem[];
  unsigned short* a_t  = (unsigned short*)smem;
  float*          ymat = (float*)(smem + (size_t)32 * XSTR * 2);
  const int row0 = blockIdx.x * 32;
  const int tid  = threadIdx.x;
  const int wv = tid >> 5, lane = tid & 31, ln = lane & 15, hf = lane >> 4;

#pragma unroll 4
  for (int i = 0; i < 32; ++i) {
    int id = tid + 256 * i;
    int r = id >> 8, c4 = (id & 255) << 2;
    float4 v = *reinterpret_cast<const float4*>(pool + (size_t)(row0 + r) * C_ + c4);
    uint2 p{pk2bf(v.x, v.y), pk2bf(v.z, v.w)};
    *reinterpret_cast<uint2*>(a_t + r * XSTR + c4) = p;
  }
  __syncthreads();

  {
    const int mt = wv & 1, ntbase = (wv >> 1) * 8;   // M tiles 2, N tiles 32
    const unsigned short* ap0 = a_t + (mt * 16 + ln) * XSTR + 16 * hf;
    // two sequential 4-tile groups keeps the double-buffer register footprint low
#pragma unroll
    for (int g = 0; g < 2; ++g) {
      const int nb = ntbase + g * 4;
      v8f acc[4] = {};
      gemm_pp<4, 32, C_>(ap0, w1T + (nb * 16 + ln) * C_ + 16 * hf, acc);
#pragma unroll
      for (int j = 0; j < 4; ++j) {
        int n = (nb + j) * 16 + ln;
#pragma unroll
        for (int r = 0; r < 8; ++r)
          ymat[(mt * 16 + r + 8 * hf) * M1_ + n] = acc[j][r];
      }
    }
  }
  __syncthreads();

  // LN over 512 + relu, 4 rows/wave
  for (int rr = 0; rr < 4; ++rr) {
    int r = wv * 4 + rr;
    float s = 0.f, s2 = 0.f;
#pragma unroll
    for (int i = 0; i < 16; ++i) {
      float v = ymat[r * M1_ + lane + 32 * i];
      s += v; s2 += v * v;
    }
    s = wred_sum(s); s2 = wred_sum(s2);
    float mu  = s * (1.f / M1_);
    float var = s2 * (1.f / M1_) - mu * mu;
    float rs  = rsqrtf(var + EPS_);
#pragma unroll
    for (int i = 0; i < 16; ++i) {
      int c = lane + 32 * i;
      float v = ymat[r * M1_ + c];
      float o = fmaxf((v - mu) * rs * g1[c] + b1[c], 0.f);
      hout[(size_t)(row0 + r) * M1_ + c] = f2bf(o);
    }
  }
}

// ---------- kernel 3: out = relu(LN(h @ W2)) -> d_out ----------
// 64 rows/block. LDS: a_t[64][HSTR] bf16 (65KB) | ymat[64][128] f32 (32KB)
__global__ void __launch_bounds__(256) k_mlp2(
    const unsigned short* __restrict__ hbf, const unsigned short* __restrict__ w2T,
    const float* __restrict__ g2, const float* __restrict__ b2,
    float* __restrict__ out) {
  extern __shared__ char smem[];
  unsigned short* a_t  = (unsigned short*)smem;
  float*          ymat = (float*)(smem + (size_t)64 * HSTR * 2);
  const int row0 = blockIdx.x * 64;
  const int tid  = threadIdx.x;
  const int wv = tid >> 5, lane = tid & 31, ln = lane & 15, hf = lane >> 4;

#pragma unroll 4
  for (int i = 0; i < 16; ++i) {
    int id = tid + 256 * i;                 // 16B chunks, 64/row
    int r = id >> 6, c8 = (id & 63) << 3;
    uint4 v = *reinterpret_cast<const uint4*>(hbf + (size_t)(row0 + r) * M1_ + c8);
    *reinterpret_cast<uint4*>(a_t + r * HSTR + c8) = v;
  }
  __syncthreads();

  {
    const int mt = wv & 3, ntbase = (wv >> 2) * 4;   // M tiles 4, N tiles 8
    v8f acc[4] = {};
    gemm_pp<4, 16, M1_>(a_t + (mt * 16 + ln) * HSTR + 16 * hf,
                        w2T + (ntbase * 16 + ln) * M1_ + 16 * hf, acc);
#pragma unroll
    for (int j = 0; j < 4; ++j) {
      int n = (ntbase + j) * 16 + ln;
#pragma unroll
      for (int r = 0; r < 8; ++r)
        ymat[(mt * 16 + r + 8 * hf) * M2_ + n] = acc[j][r];
    }
  }
  __syncthreads();

  // LN over 128 + relu, 8 rows/wave
  for (int rr = 0; rr < 8; ++rr) {
    int r = wv * 8 + rr;
    float s = 0.f, s2 = 0.f;
#pragma unroll
    for (int i = 0; i < 4; ++i) {
      float v = ymat[r * M2_ + lane + 32 * i];
      s += v; s2 += v * v;
    }
    s = wred_sum(s); s2 = wred_sum(s2);
    float mu  = s * (1.f / M2_);
    float var = s2 * (1.f / M2_) - mu * mu;
    float rs  = rsqrtf(var + EPS_);
#pragma unroll
    for (int i = 0; i < 4; ++i) {
      int c = lane + 32 * i;
      float v = ymat[r * M2_ + c];
      out[(size_t)(row0 + r) * M2_ + c] = fmaxf((v - mu) * rs * g2[c] + b2[c], 0.f);
    }
  }
}

// ---------- host ----------
extern "C" void kernel_launch(void* const* d_in, const int* in_sizes, int n_in,
                              void* d_out, int out_size, void* d_ws, size_t ws_size,
                              hipStream_t stream) {
  const float*   x   = (const float*)d_in[0];
  /* d_in[1] = temp_idx: unused by reference */
  const uint8_t* kpm = (const uint8_t*)d_in[2];
  const float*   Wk  = (const float*)d_in[3];
  const float*   bk  = (const float*)d_in[4];
  const float*   q   = (const float*)d_in[5];
  const float*   W1  = (const float*)d_in[6];
  const float*   g1  = (const float*)d_in[7];
  const float*   b1  = (const float*)d_in[8];
  const float*   W2  = (const float*)d_in[9];
  const float*   g2  = (const float*)d_in[10];
  const float*   b2  = (const float*)d_in[11];

  char* ws = (char*)d_ws;
  unsigned short* wkT  = (unsigned short*)(ws + OFF_WKT);
  unsigned short* w1T  = (unsigned short*)(ws + OFF_W1T);
  unsigned short* w2T  = (unsigned short*)(ws + OFF_W2T);
  float*          pool = (float*)(ws + OFF_POOL);
  unsigned short* hbf  = (unsigned short*)(ws + OFF_H);

  float* out  = (float*)d_out;                   // (B, M2)
  float* attn = out + (size_t)B_ * M2_;          // (B, H, T)

  const size_t S1 = (size_t)T_ * XSTR * 2 + (size_t)T_ * NK_ * 4 + (size_t)H_ * T_ * 4;
  const size_t S2 = (size_t)32 * XSTR * 2 + (size_t)32 * M1_ * 4;
  const size_t S3 = (size_t)64 * HSTR * 2 + (size_t)64 * M2_ * 4;

  k_prep<<<1024, 256, 0, stream>>>(Wk, W1, W2, wkT, w1T, w2T);
  k_attn<<<B_, 256, S1, stream>>>(x, kpm, bk, q, wkT, pool, attn);
  k_mlp1<<<B_ / 32, 256, S2, stream>>>(pool, w1T, g1, b1, hbf);
  k_mlp2<<<B_ / 64, 256, S3, stream>>>(hbf, w2T, g2, b2, out);
}
